// SoftGroupAttention_27393301414384
// MI455X (gfx1250) — matrix operand order repaired and checked by reading, not compiled, not run on hardware
//
#include <hip/hip_runtime.h>
#include <hip/hip_bf16.h>
#include <math.h>

typedef __bf16 bf16;
typedef bf16 v16bf __attribute__((ext_vector_type(16)));
typedef bf16 v8bf  __attribute__((ext_vector_type(8)));
typedef float v8f  __attribute__((ext_vector_type(8)));

constexpr int BATCH = 16;
constexpr int SEQ   = 1024;
constexpr int D     = 768;
constexpr int D3    = 2304;
constexpr int GP    = 49;
constexpr int GPP   = 64;   // padded group count
constexpr float SCALE = 0.03608439182435161f; // 768^-0.5

// ---------------- WMMA fragment helpers (CDNA5 16x16x32 bf16 layouts) ----------------
// A (16xK row-major, ld halfs): lane<16 -> kbase 0, lane>=16 -> kbase 8;
// halves [0..7] = K kbase..kbase+7, halves [8..15] = K kbase+16..kbase+23.
__device__ __forceinline__ v16bf load_a_frag(const bf16* tile, int ld, int lane) {
    int row = lane & 15;
    int kb  = (lane < 16) ? 0 : 8;
    const bf16* p = tile + (size_t)row * ld + kb;
    v8bf lo = *(const v8bf*)(p);
    v8bf hi = *(const v8bf*)(p + 16);
    return __builtin_shufflevector(lo, hi, 0,1,2,3,4,5,6,7,8,9,10,11,12,13,14,15);
}
// B given as B^T (N rows x K cols row-major, ld halfs): lane<16 -> kbase 0 else 8;
// halves [0..7] = K kbase..kbase+7, halves [8..15] = K kbase+16..kbase+23 of column (lane&15).
__device__ __forceinline__ v16bf load_bt_frag(const bf16* tileT, int ld, int lane) {
    int col = lane & 15;
    int kb  = (lane < 16) ? 0 : 8;
    const bf16* p = tileT + (size_t)col * ld + kb;
    v8bf lo = *(const v8bf*)(p);
    v8bf hi = *(const v8bf*)(p + 16);
    return __builtin_shufflevector(lo, hi, 0,1,2,3,4,5,6,7,8,9,10,11,12,13,14,15);
}
__device__ __forceinline__ v8f wmma_bf16(v16bf a, v16bf b, v8f c) {
    return __builtin_amdgcn_wmma_f32_16x16x32_bf16(false, a, false, b, (short)0, c, false, false);
}

// ---------------- prep: fp32 -> bf16 conversions / transposes ----------------
__global__ void k_cvt_x(const float* __restrict__ x, bf16* __restrict__ xb, int n) {
    for (int i = blockIdx.x * blockDim.x + threadIdx.x; i < n; i += gridDim.x * blockDim.x)
        xb[i] = (bf16)x[i];
}

__global__ void k_prep_w(const float* __restrict__ Wqkv, const float* __restrict__ Wproj,
                         const float* __restrict__ Wgp,
                         bf16* __restrict__ wqkvT, bf16* __restrict__ wprojT,
                         bf16* __restrict__ wgpT) {
    const int n1 = D * D3;        // W_qkv
    const int n2 = D * D;         // W_proj
    int total = n1 + n2 + GPP * D;
    for (int idx = blockIdx.x * blockDim.x + threadIdx.x; idx < total;
         idx += gridDim.x * blockDim.x) {
        if (idx < n1) {
            int i = idx / D3, j = idx % D3;               // W_qkv[i][j], (768 x 2304)
            wqkvT[(size_t)j * D + i] = (bf16)Wqkv[idx];
        } else if (idx < n1 + n2) {
            int t = idx - n1;
            int i = t / D, j = t % D;                     // W_proj[i][j]
            wprojT[(size_t)j * D + i] = (bf16)Wproj[t];
        } else {
            int t = idx - n1 - n2;
            int j = t / D, i = t % D;                     // wgpT[j][i], j in [0,64)
            wgpT[(size_t)j * D + i] = (j < GP) ? (bf16)Wgp[(size_t)i * GP + j] : (bf16)0.0f;
        }
    }
}

// ---------------- qkv GEMM: (16384 x 2304) = xb @ W_qkv + b ----------------
// 32x32 block per wave (2x2 tiles): 2.0 b128 loads per wmma, no spills.
__global__ void k_qkv(const bf16* __restrict__ xb, const bf16* __restrict__ wqkvT,
                      const float* __restrict__ bqkv,
                      bf16* __restrict__ q, bf16* __restrict__ k, bf16* __restrict__ vT) {
    int lane = threadIdx.x & 31;
    int wave = threadIdx.x >> 5;
    long job = (long)blockIdx.x * 8 + wave;
    const int NB = D3 / 32;                       // 72 column blocks
    int mb = (int)(job / NB);
    int nb = (int)(job % NB);
    if (mb >= (BATCH * SEQ) / 32) return;

    const bf16* A  = xb + (size_t)mb * 32 * D;
    const bf16* Bt = wqkvT + (size_t)nb * 32 * D;
    v8f acc00 = {}, acc01 = {}, acc10 = {}, acc11 = {};
    for (int kk = 0; kk < D; kk += 32) {
        v16bf a0 = load_a_frag(A + kk, D, lane);
        v16bf a1 = load_a_frag(A + (size_t)16 * D + kk, D, lane);
        v16bf b0 = load_bt_frag(Bt + kk, D, lane);
        v16bf b1 = load_bt_frag(Bt + (size_t)16 * D + kk, D, lane);
        acc00 = wmma_bf16(a0, b0, acc00);
        acc01 = wmma_bf16(a0, b1, acc01);
        acc10 = wmma_bf16(a1, b0, acc10);
        acc11 = wmma_bf16(a1, b1, acc11);
    }
    int cl = lane & 15;
    int rb = (lane < 16) ? 0 : 8;
    v8f* accs[2][2] = {{&acc00, &acc01}, {&acc10, &acc11}};
    for (int j = 0; j < 2; j++) {
        int col = nb * 32 + j * 16 + cl;
        float bias = bqkv[col];
        for (int i = 0; i < 2; i++) {
            int rbase = mb * 32 + i * 16 + rb;
            for (int e = 0; e < 8; e++) {
                int r  = rbase + e;
                int bb = r >> 10, nn = r & 1023;
                float v = (*accs[i][j])[e] + bias;
                if (col < D)            q[((size_t)bb * SEQ + nn) * D + col] = (bf16)v;
                else if (col < 2 * D)   k[((size_t)bb * SEQ + nn) * D + (col - D)] = (bf16)v;
                else                    vT[((size_t)bb * D + (col - 2 * D)) * SEQ + nn] = (bf16)v;
            }
        }
    }
}

// ---------------- gw = softmax(q @ W_gp), padded to 64 cols ----------------
__global__ void k_gw(const bf16* __restrict__ q, const bf16* __restrict__ wgpT,
                     bf16* __restrict__ gw) {
    int lane = threadIdx.x & 31;
    int wave = threadIdx.x >> 5;
    int id = blockIdx.x * 8 + wave;               // 0..1023
    if (id >= BATCH * (SEQ / 16)) return;
    int b = id >> 6, nt = id & 63;

    const bf16* A = q + ((size_t)b * SEQ + nt * 16) * D;
    v8f acc[4] = {};
    for (int kk = 0; kk < D; kk += 32) {
        v16bf a = load_a_frag(A + kk, D, lane);
        for (int t = 0; t < 4; t++) {
            v16bf bb = load_bt_frag(wgpT + (size_t)t * 16 * D + kk, D, lane);
            acc[t] = wmma_bf16(a, bb, acc[t]);
        }
    }
    int cl = lane & 15;
    int rbase = (lane < 16) ? 0 : 8;
    for (int i = 0; i < 8; i++) {
        float v[4];
        float m = -1e30f;
        for (int t = 0; t < 4; t++) {
            int cg = t * 16 + cl;
            v[t] = (cg < GP) ? acc[t][i] : -1e30f;
            m = fmaxf(m, v[t]);
        }
        for (int off = 8; off >= 1; off >>= 1) m = fmaxf(m, __shfl_xor(m, off, 16));
        float e[4]; float s = 0.0f;
        for (int t = 0; t < 4; t++) {
            int cg = t * 16 + cl;
            e[t] = (cg < GP) ? __expf(v[t] - m) : 0.0f;
            s += e[t];
        }
        for (int off = 8; off >= 1; off >>= 1) s += __shfl_xor(s, off, 16);
        float inv = 1.0f / s;
        int r = nt * 16 + rbase + i;
        for (int t = 0; t < 4; t++)
            gw[((size_t)b * SEQ + r) * GPP + t * 16 + cl] = (bf16)(e[t] * inv);
    }
}

// ---------------- pass A: row max / row sumexp over all keys ----------------
__global__ void k_stats(const bf16* __restrict__ q, const bf16* __restrict__ k,
                        float* __restrict__ rowmax, float* __restrict__ rowsum) {
    __shared__ __align__(16) bf16 qs[16 * D];
    __shared__ float Ml[8][16], Ll[8][16];
    int b = blockIdx.x >> 6, qt = blockIdx.x & 63;
    int lane = threadIdx.x & 31, wave = threadIdx.x >> 5;

    const uint4* src = (const uint4*)(q + ((size_t)b * SEQ + qt * 16) * D);
    uint4* dst = (uint4*)qs;
    for (int i = threadIdx.x; i < (16 * D) / 8; i += 256) dst[i] = src[i];
    __syncthreads();

    float m[8], l[8];
    for (int i = 0; i < 8; i++) { m[i] = -1e30f; l[i] = 0.0f; }

    for (int mt = wave; mt < 64; mt += 8) {
        const bf16* Bt = k + ((size_t)b * SEQ + mt * 16) * D;
        if (mt + 8 < 64) {  // prefetch next key tile for this wave
            const bf16* Pf = k + ((size_t)b * SEQ + (mt + 8) * 16) * D;
            __builtin_prefetch(Pf + ((size_t)(lane & 15)) * D, 0, 1);
        }
        v8f s = {};
        for (int kk = 0; kk < D; kk += 32) {
            v16bf a  = load_a_frag(qs + kk, D, lane);
            v16bf bb = load_bt_frag(Bt + kk, D, lane);
            s = wmma_bf16(a, bb, s);
        }
        for (int i = 0; i < 8; i++) {
            float sv = s[i] * SCALE;
            float tm = sv;
            for (int off = 8; off >= 1; off >>= 1) tm = fmaxf(tm, __shfl_xor(tm, off, 16));
            float mn = fmaxf(m[i], tm);
            float e = __expf(sv - mn);
            for (int off = 8; off >= 1; off >>= 1) e += __shfl_xor(e, off, 16);
            l[i] = l[i] * __expf(m[i] - mn) + e;
            m[i] = mn;
        }
    }
    int rbase = (lane < 16) ? 0 : 8;
    if ((lane & 15) == 0)
        for (int i = 0; i < 8; i++) { Ml[wave][rbase + i] = m[i]; Ll[wave][rbase + i] = l[i]; }
    __syncthreads();
    if (threadIdx.x < 16) {
        int r = threadIdx.x;
        float mg = -1e30f;
        for (int w = 0; w < 8; w++) mg = fmaxf(mg, Ml[w][r]);
        float lg = 0.0f;
        for (int w = 0; w < 8; w++) lg += Ll[w][r] * __expf(Ml[w][r] - mg);
        rowmax[(size_t)b * SEQ + qt * 16 + r] = mg;
        rowsum[(size_t)b * SEQ + qt * 16 + r] = lg;
    }
}

// ---------------- pass B: fused P = softmax*mix, O = P @ V ----------------
__global__ void k_attn_out(const bf16* __restrict__ q, const bf16* __restrict__ k,
                           const bf16* __restrict__ vT, const bf16* __restrict__ gw,
                           const float* __restrict__ rowmax, const float* __restrict__ rowsum,
                           const float* __restrict__ alpha, bf16* __restrict__ ao) {
    __shared__ __align__(16) bf16 qs[16 * D];
    __shared__ __align__(16) bf16 Ps[16 * 128];
    int b = blockIdx.x >> 6, qt = blockIdx.x & 63;
    int lane = threadIdx.x & 31, wave = threadIdx.x >> 5;

    const uint4* src = (const uint4*)(q + ((size_t)b * SEQ + qt * 16) * D);
    uint4* dst = (uint4*)qs;
    for (int i = threadIdx.x; i < (16 * D) / 8; i += 256) dst[i] = src[i];
    __syncthreads();

    float a_sig = 1.0f / (1.0f + __expf(-alpha[0]));
    int rbase = (lane < 16) ? 0 : 8;
    int cl = lane & 15;
    float rm[8], rlinv[8];
    for (int i = 0; i < 8; i++) {
        size_t ri = (size_t)b * SEQ + qt * 16 + rbase + i;
        rm[i] = rowmax[ri];
        rlinv[i] = 1.0f / rowsum[ri];
    }
    const bf16* GA = gw + ((size_t)b * SEQ + qt * 16) * GPP;
    v16bf ga0 = load_a_frag(GA, GPP, lane);
    v16bf ga1 = load_a_frag(GA + 32, GPP, lane);

    v8f acc[6] = {};
    for (int blk = 0; blk < 8; blk++) {
        int mt = blk * 8 + wave;
        const bf16* Bt = k + ((size_t)b * SEQ + mt * 16) * D;
        if (blk + 1 < 8) {  // prefetch next block's key tile for this wave
            const bf16* Pf = k + ((size_t)b * SEQ + (mt + 8) * 16) * D;
            __builtin_prefetch(Pf + ((size_t)(lane & 15)) * D, 0, 1);
        }
        // scores tile (16 q-rows x 16 keys)
        v8f s = {};
        for (int kk = 0; kk < D; kk += 32) {
            v16bf a  = load_a_frag(qs + kk, D, lane);
            v16bf bb = load_bt_frag(Bt + kk, D, lane);
            s = wmma_bf16(a, bb, s);
        }
        // group tile: GWq (16x64) @ GWk^T
        const bf16* GB = gw + ((size_t)b * SEQ + mt * 16) * GPP;
        v8f g = {};
        g = wmma_bf16(ga0, load_bt_frag(GB, GPP, lane), g);
        g = wmma_bf16(ga1, load_bt_frag(GB + 32, GPP, lane), g);

        __syncthreads();   // previous block's P fully consumed
        for (int i = 0; i < 8; i++) {
            float mix = a_sig + (1.0f - a_sig) * g[i];
            float p = __expf(s[i] * SCALE - rm[i]) * rlinv[i] * mix;
            Ps[(rbase + i) * 128 + wave * 16 + cl] = (bf16)p;
        }
        __syncthreads();
        // O += P(16x128) @ V(128 x 96-per-wave), via vT
        for (int kc = 0; kc < 4; kc++) {
            v16bf pa = load_a_frag(Ps + kc * 32, 128, lane);
            for (int t = 0; t < 6; t++) {
                int dt = wave * 6 + t;
                const bf16* Vt = vT + ((size_t)b * D + dt * 16) * SEQ + blk * 128 + kc * 32;
                acc[t] = wmma_bf16(pa, load_bt_frag(Vt, SEQ, lane), acc[t]);
            }
        }
    }
    for (int t = 0; t < 6; t++) {
        int dt = wave * 6 + t;
        for (int i = 0; i < 8; i++)
            ao[((size_t)b * SEQ + qt * 16 + rbase + i) * D + dt * 16 + cl] = (bf16)acc[t][i];
    }
}

// ---------------- final projection: out = ao @ W_proj + b_proj (fp32 out) ----------------
// 32x32 block per wave (2x2 tiles).
__global__ void k_proj(const bf16* __restrict__ ao, const bf16* __restrict__ wprojT,
                       const float* __restrict__ bproj, float* __restrict__ out) {
    int lane = threadIdx.x & 31;
    int wave = threadIdx.x >> 5;
    long job = (long)blockIdx.x * 8 + wave;
    const int NB = D / 32;                        // 24
    int mb = (int)(job / NB);
    int nb = (int)(job % NB);
    if (mb >= (BATCH * SEQ) / 32) return;

    const bf16* A  = ao + (size_t)mb * 32 * D;
    const bf16* Bt = wprojT + (size_t)nb * 32 * D;
    v8f acc00 = {}, acc01 = {}, acc10 = {}, acc11 = {};
    for (int kk = 0; kk < D; kk += 32) {
        v16bf a0 = load_a_frag(A + kk, D, lane);
        v16bf a1 = load_a_frag(A + (size_t)16 * D + kk, D, lane);
        v16bf b0 = load_bt_frag(Bt + kk, D, lane);
        v16bf b1 = load_bt_frag(Bt + (size_t)16 * D + kk, D, lane);
        acc00 = wmma_bf16(a0, b0, acc00);
        acc01 = wmma_bf16(a0, b1, acc01);
        acc10 = wmma_bf16(a1, b0, acc10);
        acc11 = wmma_bf16(a1, b1, acc11);
    }
    int cl = lane & 15;
    int rb = (lane < 16) ? 0 : 8;
    v8f* accs[2][2] = {{&acc00, &acc01}, {&acc10, &acc11}};
    for (int j = 0; j < 2; j++) {
        int col = nb * 32 + j * 16 + cl;
        float bias = bproj[col];
        for (int i = 0; i < 2; i++) {
            int rbase = mb * 32 + i * 16 + rb;
            for (int e = 0; e < 8; e++)
                out[(size_t)(rbase + e) * D + col] = (*accs[i][j])[e] + bias;
        }
    }
}

// ---------------- host launch ----------------
extern "C" void kernel_launch(void* const* d_in, const int* in_sizes, int n_in,
                              void* d_out, int out_size, void* d_ws, size_t ws_size,
                              hipStream_t stream) {
    const float* x     = (const float*)d_in[0];
    const float* Wqkv  = (const float*)d_in[1];
    const float* bqkv  = (const float*)d_in[2];
    const float* Wproj = (const float*)d_in[3];
    const float* bproj = (const float*)d_in[4];
    const float* Wgp   = (const float*)d_in[5];
    const float* alpha = (const float*)d_in[6];
    float* out = (float*)d_out;

    // workspace layout (bytes)
    char* ws = (char*)d_ws;
    size_t off = 0;
    auto alloc = [&](size_t bytes) { char* p = ws + off; off += (bytes + 255) & ~size_t(255); return p; };
    bf16* xb     = (bf16*)alloc((size_t)BATCH * SEQ * D * 2);     // x bf16 (later reused as ao)
    bf16* wqkvT  = (bf16*)alloc((size_t)D3 * D * 2);
    bf16* wprojT = (bf16*)alloc((size_t)D * D * 2);
    bf16* wgpT   = (bf16*)alloc((size_t)GPP * D * 2);
    bf16* qb     = (bf16*)alloc((size_t)BATCH * SEQ * D * 2);
    bf16* kb     = (bf16*)alloc((size_t)BATCH * SEQ * D * 2);
    bf16* vT     = (bf16*)alloc((size_t)BATCH * D * SEQ * 2);
    bf16* gw     = (bf16*)alloc((size_t)BATCH * SEQ * GPP * 2);
    float* rmax  = (float*)alloc((size_t)BATCH * SEQ * 4);
    float* rsum  = (float*)alloc((size_t)BATCH * SEQ * 4);
    bf16* ao     = xb;  // alias: xb dead after k_qkv

    const int nx = BATCH * SEQ * D;
    k_cvt_x<<<(nx + 255) / 256, 256, 0, stream>>>(x, xb, nx);

    const int nw = D * D3 + D * D + GPP * D;
    k_prep_w<<<(nw + 255) / 256, 256, 0, stream>>>(Wqkv, Wproj, Wgp, wqkvT, wprojT, wgpT);

    // qkv: 512 row-blocks * 72 col-blocks = 36864 wave jobs, 8/block
    k_qkv<<<36864 / 8, 256, 0, stream>>>(xb, wqkvT, bqkv, qb, kb, vT);

    // gw: 1024 waves, 8/block
    k_gw<<<1024 / 8, 256, 0, stream>>>(qb, wgpT, gw);

    // pass A + pass B: one block per (batch, 16-row q tile)
    k_stats<<<BATCH * (SEQ / 16), 256, 0, stream>>>(qb, kb, rmax, rsum);
    k_attn_out<<<BATCH * (SEQ / 16), 256, 0, stream>>>(qb, kb, vT, gw, rmax, rsum, alpha, ao);

    // proj: 512 row-blocks * 24 col-blocks = 12288 wave jobs, 8/block
    k_proj<<<12288 / 8, 256, 0, stream>>>(ao, wprojT, bproj, out);
}
